// DecoderRNN_67233418052069
// MI455X (gfx1250) — matrix operand order repaired and checked by reading, not compiled, not run on hardware
//
#include <hip/hip_runtime.h>
#include <hip/hip_bf16.h>

// ---------------- problem dims ----------------
#define BB 128
#define TT 17
#define NN 196
#define HH 1024
#define EE 512
#define CC 1000

typedef __attribute__((ext_vector_type(16))) __bf16 v16bf;
typedef __attribute__((ext_vector_type(8)))  __bf16 v8bf;
typedef __attribute__((ext_vector_type(8)))  float  v8f;

// ---------------------------------------------------------------
// Generic fp32 -> bf16 conversion
// ---------------------------------------------------------------
__global__ __launch_bounds__(256) void k_f32_to_bf16(const float* __restrict__ src,
                                                     __bf16* __restrict__ dst, long n) {
  long i = (long)blockIdx.x * 256 + threadIdx.x;
  if (i < n) dst[i] = (__bf16)src[i];
}

// ---------------------------------------------------------------
// Embedding + SOS prepend: emb_in[b,t,:] = t==0 ? sos : embed[labels[b,t-1]]
// ---------------------------------------------------------------
__global__ __launch_bounds__(256) void k_embed(const int* __restrict__ labels,
                                               const float* __restrict__ sos,
                                               const float* __restrict__ table,
                                               __bf16* __restrict__ emb_bf) {
  long i = (long)blockIdx.x * 256 + threadIdx.x;
  long total = (long)BB * TT * EE;
  if (i >= total) return;
  int e  = (int)(i % EE);
  long bt = i / EE;
  int t  = (int)(bt % TT);
  int b  = (int)(bt / TT);
  float v;
  if (t == 0) v = sos[e];
  else        v = table[(long)labels[b * TT + (t - 1)] * EE + e];
  emb_bf[i] = (__bf16)v;
}

// ---------------------------------------------------------------
// h init: broadcast h0 to [B,H] in fp32 and bf16
// ---------------------------------------------------------------
__global__ __launch_bounds__(256) void k_init_h(const float* __restrict__ h0,
                                                float* __restrict__ h_f,
                                                __bf16* __restrict__ h_bf) {
  long i = (long)blockIdx.x * 256 + threadIdx.x;
  if (i >= (long)BB * HH) return;
  float v = h0[i % HH];
  h_f[i]  = v;
  h_bf[i] = (__bf16)v;
}

// ---------------------------------------------------------------
// WMMA GEMM:  C[M,N] = A[M,K](bf16,row-major) * W[N,K]^T(bf16,row-major) + bias
// One wave computes a 16x64 output strip (4 WMMA tiles along N).
// M must be a multiple of 16; K a multiple of 32; N ragged-guarded.
// Output either fp32 (Cf) or bf16 (Cb).
// ---------------------------------------------------------------
__global__ __launch_bounds__(256) void k_wmma_gemm(const __bf16* __restrict__ A,
                                                   const __bf16* __restrict__ W,
                                                   const float* __restrict__ bias,
                                                   float* __restrict__ Cf,
                                                   __bf16* __restrict__ Cb,
                                                   int M, int N, int K,
                                                   int num_m_tiles, int num_n_groups) {
  const int wave = threadIdx.x >> 5;
  const int lane = threadIdx.x & 31;
  const long gw = (long)blockIdx.x * 8 + wave;
  if (gw >= (long)num_m_tiles * num_n_groups) return;

  const int mt = (int)(gw % num_m_tiles);
  const int ng = (int)(gw / num_m_tiles);
  const int mrow  = mt * 16;
  const int nbase = ng * 64;

  const int lhalf = lane >> 4;          // 0 or 1
  const int l15   = lane & 15;

  // A fragment addressing (16-bit A 16x32 layout): row = mrow + (lane&15),
  // K chunk base = 0 (lanes 0-15) or 16 (lanes 16-31); elems 0..15 -> K+0..15
  // (contiguous, matching B).
  const int arow = mrow + l15;
  const long abase = (long)arow * K + (lhalf << 4);

  // B fragment: col = nbase + j*16 + (lane&15); lanes 0-15 hold K 0..15,
  // lanes 16-31 hold K 16..31 (16 contiguous bf16 from W row).
  int wcol[4];
#pragma unroll
  for (int j = 0; j < 4; ++j) {
    int c = nbase + j * 16 + l15;
    wcol[j] = c < N ? c : (N - 1);      // clamp loads; store is guarded
  }
  const int kbofs = lhalf << 4;

  v8f acc[4] = {};

  for (int k0 = 0; k0 < K; k0 += 32) {
    const __bf16* ap = A + abase + k0;
    v8bf alo = *(const v8bf*)(ap);
    v8bf ahi = *(const v8bf*)(ap + 8);
    v16bf afrag;
#pragma unroll
    for (int i = 0; i < 8; ++i) { afrag[i] = alo[i]; afrag[i + 8] = ahi[i]; }

#pragma unroll
    for (int j = 0; j < 4; ++j) {
      const v16bf bfrag = *(const v16bf*)(W + (long)wcol[j] * K + k0 + kbofs);
      acc[j] = __builtin_amdgcn_wmma_f32_16x16x32_bf16(
          false, afrag, false, bfrag, (short)0, acc[j], false, false);
    }
  }

  // D layout: VGPR r -> row mrow + r + 8*lhalf, col = nbase + j*16 + l15
#pragma unroll
  for (int j = 0; j < 4; ++j) {
    int col = nbase + j * 16 + l15;
    if (col >= N) continue;
    float bv = bias ? bias[col] : 0.0f;
#pragma unroll
    for (int r = 0; r < 8; ++r) {
      int m = mrow + r + (lhalf << 3);
      float v = acc[j][r] + bv;
      if (Cb) Cb[(long)m * N + col] = (__bf16)v;
      else    Cf[(long)m * N + col] = v;
    }
  }
}

// ---------------------------------------------------------------
// Fused Bahdanau attention step (one workgroup per batch element):
//   scores[n] = v . tanh(cnn_proj[b,n,:] + hq[b,:]) + v_b
//   w = softmax(scores);  ctx = w @ cnn_feat[b]
//   x[b] = concat(emb_in[b,t], ctx)   (bf16)
// cnn_feat/cnn_proj are bf16 (100 MB combined -> resident in 192 MB L2).
// ---------------------------------------------------------------
__global__ __launch_bounds__(256) void k_attention(const __bf16* __restrict__ cnn_bf,
                                                   const __bf16* __restrict__ proj_bf,
                                                   const float* __restrict__ hq,
                                                   const float* __restrict__ v_w,
                                                   const float* __restrict__ v_b,
                                                   const __bf16* __restrict__ emb_bf,
                                                   __bf16* __restrict__ x_bf,
                                                   int t) {
  __shared__ float hq_s[HH];
  __shared__ float vw_s[HH];
  __shared__ float sc[256];
  __shared__ float red[256];

  const int b    = blockIdx.x;
  const int tid  = threadIdx.x;
  const int wave = tid >> 5;
  const int lane = tid & 31;

  for (int i = tid; i < HH; i += 256) {
    hq_s[i] = hq[(long)b * HH + i];
    vw_s[i] = v_w[i];
  }
  sc[tid] = -1e30f;
  __syncthreads();

  // ---- scores: one wave per attention position n ----
  const float vb = v_b[0];
  for (int n = wave; n < NN; n += 8) {
    const __bf16* row = proj_bf + ((long)b * NN + n) * HH;
    float p = 0.0f;
    for (int h = lane; h < HH; h += 32)
      p += vw_s[h] * tanhf((float)row[h] + hq_s[h]);
#pragma unroll
    for (int off = 16; off > 0; off >>= 1)
      p += __shfl_xor(p, off, 32);
    if (lane == 0) sc[n] = p + vb;
  }
  __syncthreads();

  // ---- softmax over N=196 (padded with -inf) ----
  float s = sc[tid];
  red[tid] = s;
  __syncthreads();
#pragma unroll
  for (int off = 128; off > 0; off >>= 1) {
    if (tid < off) red[tid] = fmaxf(red[tid], red[tid + off]);
    __syncthreads();
  }
  float mx = red[0];
  __syncthreads();
  float e = (tid < NN) ? __expf(s - mx) : 0.0f;
  red[tid] = e;
  __syncthreads();
#pragma unroll
  for (int off = 128; off > 0; off >>= 1) {
    if (tid < off) red[tid] += red[tid + off];
    __syncthreads();
  }
  float inv = 1.0f / red[0];
  __syncthreads();
  sc[tid] = e * inv;
  __syncthreads();

  // ---- ctx = w @ cnn_feat[b];  x = [emb_t, ctx] in bf16 ----
  for (int h = tid; h < HH; h += 256) {
    float acc = 0.0f;
    const __bf16* base = cnn_bf + (long)b * NN * HH + h;
    for (int n = 0; n < NN; ++n)
      acc += sc[n] * (float)base[(long)n * HH];
    x_bf[(long)b * (EE + HH) + EE + h] = (__bf16)acc;
  }
  for (int e2 = tid; e2 < EE; e2 += 256)
    x_bf[(long)b * (EE + HH) + e2] = emb_bf[((long)b * TT + t) * EE + e2];
}

// ---------------------------------------------------------------
// GRU elementwise update
// ---------------------------------------------------------------
__global__ __launch_bounds__(256) void k_gru(const float* __restrict__ gi,
                                             const float* __restrict__ gh,
                                             float* __restrict__ h_f,
                                             __bf16* __restrict__ h_bf,
                                             __bf16* __restrict__ outs_bf,
                                             int t) {
  long i = (long)blockIdx.x * 256 + threadIdx.x;
  if (i >= (long)BB * HH) return;
  int b = (int)(i / HH);
  int h = (int)(i % HH);
  long g = (long)b * 3 * HH + h;
  float ir = gi[g],          hr = gh[g];
  float iz = gi[g + HH],     hz = gh[g + HH];
  float in = gi[g + 2 * HH], hn = gh[g + 2 * HH];
  float r = 1.0f / (1.0f + __expf(-(ir + hr)));
  float z = 1.0f / (1.0f + __expf(-(iz + hz)));
  float n = tanhf(in + r * hn);
  float hp = h_f[i];
  float hw = (1.0f - z) * n + z * hp;
  h_f[i]  = hw;
  h_bf[i] = (__bf16)hw;
  outs_bf[((long)b * TT + t) * HH + h] = (__bf16)hw;
}

// ---------------------------------------------------------------
// Host side
// ---------------------------------------------------------------
static inline size_t align256(size_t x) { return (x + 255) & ~(size_t)255; }

static void launch_gemm(const __bf16* A, const __bf16* W, const float* bias,
                        float* Cf, __bf16* Cb, int M, int N, int K,
                        hipStream_t s) {
  int mt = M / 16;
  int ng = (N + 63) / 64;
  long total = (long)mt * ng;
  int blocks = (int)((total + 7) / 8);
  k_wmma_gemm<<<blocks, 256, 0, s>>>(A, W, bias, Cf, Cb, M, N, K, mt, ng);
}

extern "C" void kernel_launch(void* const* d_in, const int* in_sizes, int n_in,
                              void* d_out, int out_size, void* d_ws, size_t ws_size,
                              hipStream_t stream) {
  const float* cnn_feat = (const float*)d_in[0];
  const int*   labels   = (const int*)d_in[1];
  // d_in[2] = lens (unused, matching reference)
  const float* sos      = (const float*)d_in[3];
  const float* h0       = (const float*)d_in[4];
  const float* table    = (const float*)d_in[5];
  const float* W_ih     = (const float*)d_in[6];
  const float* b_ih     = (const float*)d_in[7];
  const float* W_hh     = (const float*)d_in[8];
  const float* b_hh     = (const float*)d_in[9];
  const float* Wh       = (const float*)d_in[10];
  const float* bh       = (const float*)d_in[11];
  const float* Wc       = (const float*)d_in[12];
  const float* bc       = (const float*)d_in[13];
  const float* v_w      = (const float*)d_in[14];
  const float* v_b      = (const float*)d_in[15];
  const float* Wcls     = (const float*)d_in[16];
  const float* bcls     = (const float*)d_in[17];
  float* out = (float*)d_out;

  // ---- workspace carve-up ----
  char* p = (char*)d_ws;
  size_t off = 0;
  auto carve = [&](size_t bytes) { void* r = p + off; off = align256(off + bytes); return r; };

  const long nCNN  = (long)BB * NN * HH;
  __bf16* cnn_bf  = (__bf16*)carve(nCNN * 2);
  __bf16* proj_bf = (__bf16*)carve(nCNN * 2);
  __bf16* wc_bf   = (__bf16*)carve((long)HH * HH * 2);
  __bf16* wh_bf   = (__bf16*)carve((long)HH * HH * 2);
  __bf16* whh_bf  = (__bf16*)carve((long)3 * HH * HH * 2);
  __bf16* wih_bf  = (__bf16*)carve((long)3 * HH * (EE + HH) * 2);
  __bf16* wcls_bf = (__bf16*)carve((long)CC * HH * 2);
  __bf16* emb_bf  = (__bf16*)carve((long)BB * TT * EE * 2);
  float*  h_f     = (float*)carve((long)BB * HH * 4);
  __bf16* h_bf    = (__bf16*)carve((long)BB * HH * 2);
  float*  hq_f    = (float*)carve((long)BB * HH * 4);
  float*  gh_f    = (float*)carve((long)BB * 3 * HH * 4);
  __bf16* x_bf    = (__bf16*)carve((long)BB * (EE + HH) * 2);
  float*  gi_f    = (float*)carve((long)BB * 3 * HH * 4);
  __bf16* outs_bf = (__bf16*)carve((long)BB * TT * HH * 2);
  (void)ws_size; (void)n_in; (void)in_sizes; (void)out_size;

  auto cvt = [&](const float* src, __bf16* dst, long n) {
    k_f32_to_bf16<<<(int)((n + 255) / 256), 256, 0, stream>>>(src, dst, n);
  };

  // ---- one-time conversions (per call; deterministic) ----
  cvt(cnn_feat, cnn_bf,  nCNN);
  cvt(Wc,       wc_bf,   (long)HH * HH);
  cvt(Wh,       wh_bf,   (long)HH * HH);
  cvt(W_hh,     whh_bf,  (long)3 * HH * HH);
  cvt(W_ih,     wih_bf,  (long)3 * HH * (EE + HH));
  cvt(Wcls,     wcls_bf, (long)CC * HH);

  {
    long n = (long)BB * TT * EE;
    k_embed<<<(int)((n + 255) / 256), 256, 0, stream>>>(labels, sos, table, emb_bf);
  }
  {
    long n = (long)BB * HH;
    k_init_h<<<(int)((n + 255) / 256), 256, 0, stream>>>(h0, h_f, h_bf);
  }

  // ---- hoisted projection: cnn_proj = cnn_feat @ Wc^T + bc  (bf16 out) ----
  launch_gemm(cnn_bf, wc_bf, bc, nullptr, proj_bf, BB * NN, HH, HH, stream);

  // ---- 17-step sequential scan ----
  const int nBH = (BB * HH + 255) / 256;
  for (int t = 0; t < TT; ++t) {
    launch_gemm(h_bf, wh_bf,  bh,   hq_f, nullptr, BB, HH,     HH, stream);      // hq
    launch_gemm(h_bf, whh_bf, b_hh, gh_f, nullptr, BB, 3 * HH, HH, stream);      // gh
    k_attention<<<BB, 256, 0, stream>>>(cnn_bf, proj_bf, hq_f, v_w, v_b,
                                        emb_bf, x_bf, t);
    launch_gemm(x_bf, wih_bf, b_ih, gi_f, nullptr, BB, 3 * HH, EE + HH, stream); // gi
    k_gru<<<nBH, 256, 0, stream>>>(gi_f, gh_f, h_f, h_bf, outs_bf, t);
  }

  // ---- classifier: logits = outs @ Wcls^T + bcls (fp32 out) ----
  launch_gemm(outs_bf, wcls_bf, bcls, out, nullptr, BB * TT, CC, HH, stream);
}
